// ImprovedGCN_6665789243727
// MI455X (gfx1250) — compile-verified, matrix operand order checked by reading, not run on hardware
//
#include <hip/hip_runtime.h>
#include <hip/hip_bf16.h>

typedef __attribute__((ext_vector_type(16))) _Float16 v16h;
typedef __attribute__((ext_vector_type(8)))  float    v8f;

#define N_NODES 100000
#define N_EDGES 1600000
#define K_DIM   128          // input feature dim of every GEMM here

// ---- CDNA5 16-bit A-matrix 16x32 K interleave (ISA 7.12.2) ----
// lane L: row M = L%16, half = L/16.
// VGPR v in 0..3 holds K = half*8 + 2v, +1 ; v in 4..7 holds K = 16 + half*8 + 2(v-4), +1
__device__ __forceinline__ int kk_of(int v, int half) {
  return (v < 4) ? (half * 8 + 2 * v) : (16 + half * 8 + 2 * (v - 4));
}

// ---------------- degree / normalization ----------------
__global__ void gcn_fill_ones(float* __restrict__ deg) {
  int i = blockIdx.x * blockDim.x + threadIdx.x;
  if (i < N_NODES) deg[i] = 1.0f;            // self-loop counts once
}

__global__ void gcn_count_deg(const int* __restrict__ dsts, float* __restrict__ deg) {
  int e = blockIdx.x * blockDim.x + threadIdx.x;
  if (e < N_EDGES) unsafeAtomicAdd(&deg[dsts[e]], 1.0f);   // global_atomic_add_f32
}

__global__ void gcn_make_dinv(const float* __restrict__ deg, float* __restrict__ dinv) {
  int i = blockIdx.x * blockDim.x + threadIdx.x;
  if (i < N_NODES) dinv[i] = rsqrtf(deg[i]); // deg >= 1 always
}

// ---------------- weight packing into WMMA B fragments ----------------
// B (K x F, fp32 row-major) -> per (ktile, ctile, lane) v16h fragment, contiguous 32B.
// B 32x16 f16 layout mirrors A: lane L holds column N = L%16, same K interleave.
__global__ void gcn_pack_B(const float* __restrict__ W, _Float16* __restrict__ bp, int F) {
  int t = blockIdx.x * blockDim.x + threadIdx.x;
  int ctiles = F / 16;
  int total  = 4 * ctiles * 32;
  if (t >= total) return;
  int lane = t & 31;
  int tile = t >> 5;
  int ct   = tile % ctiles;
  int kt   = tile / ctiles;
  int n    = lane & 15;
  int half = lane >> 4;
  _Float16* o = bp + (size_t)t * 16;
  #pragma unroll
  for (int v = 0; v < 8; ++v) {
    int k = kt * 32 + kk_of(v, half);
    o[2 * v]     = (_Float16)W[(size_t)k       * F + ct * 16 + n];
    o[2 * v + 1] = (_Float16)W[(size_t)(k + 1) * F + ct * 16 + n];
  }
}

// ---------------- WMMA GEMM: out[N x F] = A[N x 128] * W ----------------
// One wave -> 16 rows x 64 cols (4 WMMA column tiles, A fragment reused 4x).
template <int F>
__global__ void gcn_gemm_wmma(const float* __restrict__ A, const _Float16* __restrict__ bp,
                              float* __restrict__ out, int n_rows) {
  const int lane = threadIdx.x & 31;
  const int wave = threadIdx.x >> 5;
  int row_tile, colbase;
  if (F == 128) {                 // 8 waves: 4 row tiles x 2 column halves
    row_tile = blockIdx.x * 4 + (wave >> 1);
    colbase  = (wave & 1) * 64;
  } else {                        // F==64: 8 waves = 8 row tiles, full width
    row_tile = blockIdx.x * 8 + wave;
    colbase  = 0;
  }
  if (row_tile * 16 >= n_rows) return;       // wave-uniform: EXEC stays all-1s for WMMA
  const int rowbase = row_tile * 16;
  const int n    = lane & 15;
  const int half = lane >> 4;
  const int r    = rowbase + n;
  const int ctiles = F / 16;

  v8f z = {0.f, 0.f, 0.f, 0.f, 0.f, 0.f, 0.f, 0.f};
  v8f c[4] = {z, z, z, z};

  for (int kt = 0; kt < 4; ++kt) {
    v16h a;
    const float* arow = A + (size_t)r * K_DIM + kt * 32;
    #pragma unroll
    for (int v = 0; v < 8; ++v) {
      int kk = kk_of(v, half);               // even -> 8B-aligned float2
      float2 f = *(const float2*)(arow + kk);
      a[2 * v]     = (_Float16)f.x;
      a[2 * v + 1] = (_Float16)f.y;
    }
    #pragma unroll
    for (int ct = 0; ct < 4; ++ct) {
      int ctg = colbase / 16 + ct;
      v16h b = *(const v16h*)(bp + ((size_t)(kt * ctiles + ctg) * 32 + lane) * 16);
      c[ct] = __builtin_amdgcn_wmma_f32_16x16x32_f16(
          false, a, false, b, (short)0, c[ct], false, false);
    }
  }
  // C/D layout: lane -> col n, VGPR v -> row v (+8 for upper half-wave)
  #pragma unroll
  for (int ct = 0; ct < 4; ++ct) {
    int cb = colbase + ct * 16 + n;
    #pragma unroll
    for (int v = 0; v < 8; ++v) {
      int m = v + half * 8;
      out[(size_t)(rowbase + m) * F + cb] = c[ct][v];
    }
  }
}

// ---------------- self-loop + bias accumulator init ----------------
template <int F>
__global__ void gcn_init_acc(const float* __restrict__ hw, const float* __restrict__ dinv,
                             const float* __restrict__ bias, float* __restrict__ acc) {
  long long i = (long long)blockIdx.x * blockDim.x + threadIdx.x;
  if (i >= (long long)N_NODES * F) return;
  int node = (int)(i / F);
  int f    = (int)(i % F);
  float di = dinv[node];
  acc[i] = bias[f] + hw[i] * di * di;        // self-loop norm = dinv^2
}

// ---------------- edge gather-scale-scatter (L2-resident) ----------------
template <int F>
__global__ void gcn_scatter(const int* __restrict__ srcs, const int* __restrict__ dsts,
                            const float* __restrict__ dinv, const float* __restrict__ hw,
                            float* __restrict__ acc) {
  const int lanes_per_edge = F / 4;
  long long gid = (long long)blockIdx.x * blockDim.x + threadIdx.x;
  long long e   = gid / lanes_per_edge;
  int       l   = (int)(gid % lanes_per_edge);
  if (e >= N_EDGES) return;
  int s = srcs[e], d = dsts[e];
  float norm = dinv[s] * dinv[d];
  const float4 v = *(const float4*)(hw + (size_t)s * F + l * 4);
  float* dp = acc + (size_t)d * F + l * 4;
  unsafeAtomicAdd(dp + 0, v.x * norm);
  unsafeAtomicAdd(dp + 1, v.y * norm);
  unsafeAtomicAdd(dp + 2, v.z * norm);
  unsafeAtomicAdd(dp + 3, v.w * norm);
}

// ---------------- elementwise ----------------
__global__ void gcn_relu(const float* __restrict__ a, float* __restrict__ o, long long cnt) {
  long long i = (long long)blockIdx.x * blockDim.x + threadIdx.x;
  if (i < cnt) o[i] = fmaxf(a[i], 0.0f);
}

__global__ void gcn_combine_skip(const float* __restrict__ acc, const float* __restrict__ skip,
                                 const float* __restrict__ bs, float* __restrict__ h) {
  long long i = (long long)blockIdx.x * blockDim.x + threadIdx.x;
  if (i >= (long long)N_NODES * 128) return;
  h[i] = fmaxf(acc[i], 0.0f) + skip[i] + bs[i & 127];
}

// ---------------- orchestration ----------------
extern "C" void kernel_launch(void* const* d_in, const int* in_sizes, int n_in,
                              void* d_out, int out_size, void* d_ws, size_t ws_size,
                              hipStream_t stream) {
  const float* x  = (const float*)d_in[0];
  const int*   ei = (const int*)d_in[1];
  const float* W0 = (const float*)d_in[2];
  const float* b0 = (const float*)d_in[3];
  const float* W1 = (const float*)d_in[4];
  const float* b1 = (const float*)d_in[5];
  const float* W2 = (const float*)d_in[6];
  const float* b2 = (const float*)d_in[7];
  const float* Wsk = (const float*)d_in[8];
  const float* bs  = (const float*)d_in[9];
  float* out = (float*)d_out;

  const int* src = ei;            // edge_index[0]
  const int* dst = ei + N_EDGES;  // edge_index[1]

  // workspace layout (~154 MB)
  float* ws   = (float*)d_ws;
  float* dinv = ws;                                    // 512KB slot
  float* h    = ws + 131072;                           // N*128
  float* hw   = h  + (size_t)N_NODES * 128;            // N*128
  float* acc  = hw + (size_t)N_NODES * 128;            // N*128
  _Float16* bpack = (_Float16*)(acc + (size_t)N_NODES * 128);  // 32KB

  const int T = 256;
  const long long NF128 = (long long)N_NODES * 128;
  const long long NF64  = (long long)N_NODES * 64;
  const int gNode   = (N_NODES + T - 1) / T;
  const int gEdge   = (N_EDGES + T - 1) / T;
  const int gNF128  = (int)((NF128 + T - 1) / T);
  const int gNF64   = (int)((NF64 + T - 1) / T);
  const int gScat128 = (int)(((long long)N_EDGES * 32 + T - 1) / T);
  const int gScat64  = (int)(((long long)N_EDGES * 16 + T - 1) / T);
  const int gGemm128 = (N_NODES / 16 + 3) / 4;   // 4 row-tiles per block
  const int gGemm64  = (N_NODES / 16 + 7) / 8;   // 8 row-tiles per block

  // degrees -> dinv (deg staged in hw, overwritten later)
  gcn_fill_ones<<<gNode, T, 0, stream>>>(hw);
  gcn_count_deg<<<gEdge, T, 0, stream>>>(dst, hw);
  gcn_make_dinv<<<gNode, T, 0, stream>>>(hw, dinv);

  // ---- layer 0: h = relu(conv(x, W0, b0)) ----
  gcn_pack_B<<<4, T, 0, stream>>>(W0, bpack, 128);
  gcn_gemm_wmma<128><<<gGemm128, T, 0, stream>>>(x, bpack, hw, N_NODES);
  gcn_init_acc<128><<<gNF128, T, 0, stream>>>(hw, dinv, b0, acc);
  gcn_scatter<128><<<gScat128, T, 0, stream>>>(src, dst, dinv, hw, acc);
  gcn_relu<<<gNF128, T, 0, stream>>>(acc, h, NF128);

  // ---- layer 1: h = relu(conv(h, W1, b1)) + h@Ws + bs ----
  gcn_pack_B<<<4, T, 0, stream>>>(W1, bpack, 128);
  gcn_gemm_wmma<128><<<gGemm128, T, 0, stream>>>(h, bpack, hw, N_NODES);
  gcn_init_acc<128><<<gNF128, T, 0, stream>>>(hw, dinv, b1, acc);
  gcn_scatter<128><<<gScat128, T, 0, stream>>>(src, dst, dinv, hw, acc);
  gcn_pack_B<<<4, T, 0, stream>>>(Wsk, bpack, 128);
  gcn_gemm_wmma<128><<<gGemm128, T, 0, stream>>>(h, bpack, hw, N_NODES); // skip into hw
  gcn_combine_skip<<<gNF128, T, 0, stream>>>(acc, hw, bs, h);

  // ---- output layer: out = conv(h, W2, b2), F=64, directly into d_out ----
  gcn_pack_B<<<2, T, 0, stream>>>(W2, bpack, 64);
  gcn_gemm_wmma<64><<<gGemm64, T, 0, stream>>>(h, bpack, hw, N_NODES);
  gcn_init_acc<64><<<gNF64, T, 0, stream>>>(hw, dinv, b2, out);
  gcn_scatter<64><<<gScat64, T, 0, stream>>>(src, dst, dinv, hw, out);
}
